// Registration3d_15874199126627
// MI455X (gfx1250) — matrix operand order repaired and checked by reading, not compiled, not run on hardware
//
#include <hip/hip_runtime.h>

typedef __attribute__((ext_vector_type(16))) _Float16 v16h;
typedef __attribute__((ext_vector_type(8)))  float    v8f;

typedef __attribute__((address_space(1))) int gas_int;  // global
typedef __attribute__((address_space(3))) int las_int;  // LDS

#define D_    64
#define H_    256
#define W_    256
#define HW_   (H_ * W_)
#define DHW_  (D_ * HW_)
#define PW_   (W_ + 2)
#define PHW_  ((H_ + 2) * (W_ + 2))
#define KTOT  54            // INC * 27
#define SLABX 132           // 128 voxels + 2 halo + 2 pad (zeroed)
#define SLABN (2 * 3 * 3 * SLABX)   // 2376 floats

// gfx1250 async global->LDS path (ASYNCcnt), with branchless fallback
#if __has_builtin(__builtin_amdgcn_global_load_async_to_lds_b32) && \
    __has_builtin(__builtin_amdgcn_s_wait_asynccnt)
#define USE_ASYNC_LDS 1
#else
#define USE_ASYNC_LDS 0
#endif

// One fused kernel: implicit-GEMM offset conv (WMMA) + trilinear gather + passthrough.
// Block = 256 threads = 8 wave32; each wave owns 16 consecutive voxels along W.
__global__ __launch_bounds__(256) void reg3d_fused(
    const float* __restrict__ x,    // (B, 2, D, H, W)
    const float* __restrict__ w_p,  // (3, 2, 3, 3, 3) -> [n*54 + k]
    const float* __restrict__ b_p,  // (3,)
    float* __restrict__ out)        // (B, 2, D, H, W)
{
  __shared__ float xs[SLABN];                      // im2col halo slab (f32)
  __shared__ __align__(32) _Float16 wlds[16 * 64]; // zero-padded weights (N=16, K=64)
  __shared__ float offs[8][16][3];                 // [wave][voxel][z,y,x offset]

  const int tid  = threadIdx.x;
  const int lane = tid & 31;
  const int wv   = tid >> 5;

  // blockIdx.x -> (b, d, h, wblk); grid = 2*64*256*2 = 65536
  const int bid  = blockIdx.x;
  const int wblk = bid & 1;
  const int h    = (bid >> 1) & (H_ - 1);
  const int d    = (bid >> 9) & (D_ - 1);
  const int b    = bid >> 15;
  const int w0   = wblk * 128 + wv * 16;

  // ---------- phase 1a: weights -> LDS, zero-padded to 16x64 (branchless) ----------
  for (int i = tid; i < 16 * 64; i += 256) {
    const int n = i >> 6, K = i & 63;
    const bool vld = (n < 3) && (K < KTOT);
    const float t = w_p[vld ? n * KTOT + K : 0];   // unconditional load, clamped addr
    wlds[i] = (_Float16)(vld ? t : 0.f);
  }

  // ---------- phase 1b: x halo slab -> LDS ----------
  for (int i = tid; i < SLABN; i += 256) {
    int t = i;
    const int xi = t % SLABX; t /= SLABX;
    const int ky = t % 3;     t /= 3;
    const int kz = t % 3;
    const int c  = t / 3;
    const int zz = d + kz - 1, yy = h + ky - 1, xx = wblk * 128 + xi - 1;
    const bool inb = ((unsigned)zz < (unsigned)D_) && ((unsigned)yy < (unsigned)H_) &&
                     ((unsigned)xx < (unsigned)W_) && (xi < 130);
    const size_t gidx = ((size_t)(b * 2 + c) * D_ + (inb ? zz : 0)) * HW_ +
                        (size_t)(inb ? yy : 0) * W_ + (inb ? xx : 0);
#if USE_ASYNC_LDS
    if (inb) {
      __builtin_amdgcn_global_load_async_to_lds_b32(
          (gas_int*)(x + gidx), (las_int*)&xs[i], 0, 0);
    } else {
      xs[i] = 0.f;   // disjoint addresses from async writes -> no hazard
    }
#else
    const float t2 = x[gidx];                      // unconditional load, clamped addr
    xs[i] = inb ? t2 : 0.f;
#endif
  }
#if USE_ASYNC_LDS
  __builtin_amdgcn_s_wait_asynccnt(0);
#endif
  __syncthreads();

  // ---------- phase 2: implicit-GEMM offset conv (two v_wmma_f32_16x16x32_f16) ----------
  const int half = lane >> 4;     // lane group (ISA 7.12.2 layouts)
  const int n    = lane & 15;     // output channel for B/C/D frags
  const int m    = lane & 15;     // A-frag row = voxel within tile
  const int bm   = wv * 16 + m;   // voxel x-position within slab (before +kx)

  v8f acc = {};
  #pragma unroll
  for (int kw = 0; kw < 2; ++kw) {
    const int kbase = kw * 32;

    // B fragment: one aligned 32B vector load from padded weight tile
    const v16h bf = *(const v16h*)&wlds[n * 64 + kbase + half * 16];

    // A fragment: branchless gather from LDS slab in contiguous k order (k = kbase + 16*half + e, same as B)
    v16h af;
    #pragma unroll
    for (int e = 0; e < 16; ++e) {
      const int j = e >> 1;
      const int K = kbase + (half << 4) + e;
      const bool vK = K < KTOT;
      const int  Kc = vK ? K : 0;
      const int  c  = Kc >= 27 ? 1 : 0;
      int        r  = Kc - c * 27;
      const int  kz = r / 9;  r -= kz * 9;
      const int  ky = r / 3;
      const int  kx = r - ky * 3;
      const float fv = xs[((c * 3 + kz) * 3 + ky) * SLABX + bm + kx];
      af[e] = (_Float16)(vK ? fv : 0.f);
    }

    acc = __builtin_amdgcn_wmma_f32_16x16x32_f16(false, af, false, bf,
                                                 (short)0, acc, false, false);
  }

  // C/D layout: VGPR r -> (M = half*8 + r, N = lane&15)
  if (n < 3) {
    const float bias = b_p[n];
    #pragma unroll
    for (int r = 0; r < 8; ++r)
      offs[wv][half * 8 + r][n] = acc[r] + bias;
  }
  __syncthreads();

  // ---------- phase 3: trilinear gather (lanes 0-15) + passthrough (lanes 16-31) ----------
  if (lane < 16) {
    const int vx  = lane;
    const int wxv = w0 + vx;
    const float oz = offs[wv][vx][0];
    const float oy = offs[wv][vx][1];
    const float ox = offs[wv][vx][2];

    // 1-based base grid; clamp to [0, size]
    const float qz = fminf(fmaxf((float)(d + 1)   + oz, 0.f), (float)D_);
    const float qy = fminf(fmaxf((float)(h + 1)   + oy, 0.f), (float)H_);
    const float qx = fminf(fmaxf((float)(wxv + 1) + ox, 0.f), (float)W_);

    const float wz1 = qz - floorf(qz);
    const float wy1 = qy - floorf(qy);
    const float wx1 = qx - floorf(qx);

    const float* xb = x + (size_t)(b * 2) * DHW_;  // moving channel (c=0)
    float sum = 0.f;
    #pragma unroll
    for (int tap = 0; tap < 8; ++tap) {
      const int dz = tap >> 2, dy = (tap >> 1) & 1, dx = tap & 1;
      const float cz = qz + (float)dz;
      const float cy = qy + (float)dy;
      const float cx = qx + (float)dx;
      // faithful to reference: combined f32 flat index, THEN truncate
      const float fidx = __fadd_rn(
          __fadd_rn(__fmul_rn(cz, (float)PHW_), __fmul_rn(cy, (float)PW_)), cx);
      const int idx = (int)fidx;
      const int z   = idx / PHW_;
      int       rr  = idx - z * PHW_;
      const int y   = rr / PW_;
      const int xxp = rr - y * PW_;
      const int zi = z - 1, yi = y - 1, xi = xxp - 1;
      const bool inb = ((unsigned)zi < (unsigned)D_) && ((unsigned)yi < (unsigned)H_) &&
                       ((unsigned)xi < (unsigned)W_);
      // unconditional load from clamped address; zero the WEIGHT when OOB
      const int zc = inb ? zi : 0, yc = inb ? yi : 0, xc = inb ? xi : 0;
      const float val = xb[(size_t)zc * HW_ + (size_t)yc * W_ + xc];
      const float wt  = (dz ? wz1 : 1.f - wz1) *
                        (dy ? wy1 : 1.f - wy1) *
                        (dx ? wx1 : 1.f - wx1);
      sum += val * (inb ? wt : 0.f);
    }
    out[(size_t)(b * 2) * DHW_ + (size_t)d * HW_ + (size_t)h * W_ + wxv] = sum;
  } else {
    const int i = lane - 16;
    const size_t off = (size_t)(b * 2 + 1) * DHW_ + (size_t)d * HW_ +
                       (size_t)h * W_ + (w0 + i);
    out[off] = x[off];
  }
}

extern "C" void kernel_launch(void* const* d_in, const int* in_sizes, int n_in,
                              void* d_out, int out_size, void* d_ws, size_t ws_size,
                              hipStream_t stream) {
  (void)in_sizes; (void)n_in; (void)out_size; (void)d_ws; (void)ws_size;
  const float* x   = (const float*)d_in[0];
  const float* w_p = (const float*)d_in[1];
  const float* b_p = (const float*)d_in[2];
  float* out = (float*)d_out;

  // B * D * H * (W/128) = 2 * 64 * 256 * 2 = 65536 blocks of 256 threads (8 wave32)
  reg3d_fused<<<dim3(65536), dim3(256), 0, stream>>>(x, w_p, b_p, out);
}
